// KeypointSelector_50345606644323
// MI455X (gfx1250) — matrix operand order repaired and checked by reading, not compiled, not run on hardware
//
#include <hip/hip_runtime.h>
#include <hip/hip_bf16.h>

// ---------------------------------------------------------------------------
// CDNA5 (gfx1250) implementation: conv3x3 layers as implicit GEMM on the
// bf16 WMMA pipe (v_wmma_f32_16x16x32_bf16), BN+ReLU fused in epilogue.
// ---------------------------------------------------------------------------

typedef __attribute__((ext_vector_type(16))) __bf16 v16bf;
typedef __attribute__((ext_vector_type(8)))  float  v8f;

#define BB   8
#define HH   128
#define WW   128
#define CIN0 384
#define HD   256
#define NPIX (BB*HH*WW)          // 131072

__device__ __forceinline__ unsigned short f2bf(float f) {
  unsigned int u = __builtin_bit_cast(unsigned int, f);
  unsigned int r = (u + 0x7FFFu + ((u >> 16) & 1u)) >> 16;   // RNE
  return (unsigned short)r;
}
__device__ __forceinline__ float bf2f(unsigned short h) {
  unsigned int u = ((unsigned int)h) << 16;
  return __builtin_bit_cast(float, u);
}

// ---------------- prep kernels ----------------

__global__ __launch_bounds__(256)
void cvt_bf16_kernel(const float* __restrict__ s, unsigned short* __restrict__ d, int n) {
  int i = blockIdx.x * 256 + threadIdx.x;
  if (i < n) d[i] = f2bf(s[i]);
}

// w: [K][256] f32 row-major (HWIO flattened)  ->  bt: [256][K] bf16 (N-major)
__global__ __launch_bounds__(256)
void transpose_w_kernel(const float* __restrict__ w, unsigned short* __restrict__ bt, int K) {
  int i = blockIdx.x * 256 + threadIdx.x;
  if (i < K * 256) {
    int k = i >> 8, n = i & 255;
    bt[n * K + k] = f2bf(w[i]);
  }
}

// fold conv bias + BN into per-channel scale/shift
__global__ __launch_bounds__(256)
void bn_fold_kernel(const float* __restrict__ bias, const float* __restrict__ g,
                    const float* __restrict__ beta, const float* __restrict__ m,
                    const float* __restrict__ v, float* __restrict__ a, float* __restrict__ c) {
  int n = threadIdx.x;
  float s = g[n] * rsqrtf(v[n] + 1e-5f);
  a[n] = s;
  c[n] = (bias[n] - m[n]) * s + beta[n];
}

// ---------------- implicit-GEMM conv3x3 + BN + ReLU (WMMA bf16) ----------------
// X   : bf16 NHWC activations, Cin channels
// Bt  : bf16 [256][K] weights, K = 9*Cin (tap-major rows of im2col), N-major
// Out : bf16 NHWC, 256 channels, ReLU(BN(conv))
// Block = 256 threads (8 waves); tile = 32 pixels (M) x 256 channels (N).
__global__ __launch_bounds__(256)
void conv_gemm_wmma(const unsigned short* __restrict__ X,
                    const unsigned short* __restrict__ Bt,
                    const float* __restrict__ scale,
                    const float* __restrict__ shift,
                    unsigned short* __restrict__ Out,
                    int Cin) {
  const int K = 9 * Cin;
  __shared__ __align__(16) unsigned short Alds[32 * 32];    // [Mrow][k]  2 KB
  __shared__ __align__(16) unsigned short Blds[256 * 32];   // [N][k]    16 KB

  const int t     = threadIdx.x;
  const int lane  = t & 31;
  const int wave  = t >> 5;       // 0..7
  const int wm    = wave >> 2;    // 0..1  (16-row slab)
  const int wn    = wave & 3;     // 0..3  (64-col slab)
  const int pBase = blockIdx.x * 32;

  v8f acc[4];
  v8f zero = {0.f,0.f,0.f,0.f,0.f,0.f,0.f,0.f};
#pragma unroll
  for (int j = 0; j < 4; ++j) acc[j] = zero;

  // A staging coordinates: thread -> (row 0..31, 4 contiguous k's)
  const int ar = t >> 3;          // 0..31
  const int ac = (t & 7) * 4;     // 0,4,...,28
  const int p  = pBase + ar;
  const int bi = p >> 14;                // / (H*W)
  const int py = (p >> 7) & (HH - 1);
  const int px = p & (WW - 1);

  for (int k0 = 0; k0 < K; k0 += 32) {
    // ---- stage A tile: im2col gather (one conv tap per 32-K step) ----
    const int tap = k0 / Cin;                  // 32 | Cin  ->  constant per step
    const int dy  = tap / 3 - 1, dx = tap % 3 - 1;
    const int yy  = py + dy, xx = px + dx;
    unsigned long long aval = 0ull;
    if ((unsigned)yy < (unsigned)HH && (unsigned)xx < (unsigned)WW) {
      const int cbase = (k0 - tap * Cin) + ac;
      aval = *reinterpret_cast<const unsigned long long*>(
                 X + (((bi * HH + yy) * WW + xx) * Cin + cbase));
    }
    *reinterpret_cast<unsigned long long*>(&Alds[ar * 32 + ac]) = aval;

    // ---- stage B tile: row n = t, 32 k's (64 B) ----
    {
      const uint4* s4 = reinterpret_cast<const uint4*>(Bt + t * K + k0);
      uint4* d4 = reinterpret_cast<uint4*>(&Blds[t * 32]);
      d4[0] = s4[0]; d4[1] = s4[1]; d4[2] = s4[2]; d4[3] = s4[3];
      if (k0 + 32 < K)
        __builtin_prefetch(Bt + t * K + k0 + 32, 0, 1);   // global_prefetch_b8
    }
    __syncthreads();

    // ---- A fragment (ISA 16-bit A 16x32 layout) ----
    union FA { v16bf v; uint4 q[2]; } fa;
    const int am  = wm * 16 + (lane & 15);
    const int akb = (lane >> 4) * 8;            // lanes>=16 start at K=8
    fa.q[0] = *reinterpret_cast<const uint4*>(&Alds[am * 32 + akb]);       // K kb..kb+7
    fa.q[1] = *reinterpret_cast<const uint4*>(&Alds[am * 32 + akb + 16]);  // K kb+16..kb+23

#pragma unroll
    for (int j = 0; j < 4; ++j) {
      // ---- B fragment (K x 16, lanes>=16 hold K=8..15,24..31) ----
      union FB { v16bf v; uint4 q[2]; } fb;
      const int bn  = wn * 64 + j * 16 + (lane & 15);
      const int bkb = (lane >> 4) * 8;
      fb.q[0] = *reinterpret_cast<const uint4*>(&Blds[bn * 32 + bkb]);
      fb.q[1] = *reinterpret_cast<const uint4*>(&Blds[bn * 32 + bkb + 16]);

      acc[j] = __builtin_amdgcn_wmma_f32_16x16x32_bf16(
                   false, fa.v, false, fb.v, (short)0, acc[j], false, false);
    }
    __syncthreads();
  }

  // ---- epilogue: fused bias/BN (scale,shift) + ReLU, emit bf16 NHWC ----
#pragma unroll
  for (int j = 0; j < 4; ++j) {
    const int n = wn * 64 + j * 16 + (lane & 15);
    const float a = scale[n], c = shift[n];
#pragma unroll
    for (int r = 0; r < 8; ++r) {
      const int ml = (lane >> 4) * 8 + r;       // C/D layout: lanes>=16 hold M=8..15
      const int pp = pBase + wm * 16 + ml;
      float v = acc[j][r] * a + c;
      v = v > 0.f ? v : 0.f;
      Out[pp * HD + n] = f2bf(v);
    }
  }
}

// ---------------- conv2 (256 -> 1) + sigmoid ----------------
__global__ __launch_bounds__(256)
void conv2_sigmoid_kernel(const unsigned short* __restrict__ act,
                          const float* __restrict__ w2, const float* __restrict__ b2,
                          float* __restrict__ sal) {
  __shared__ float wl[9 * HD];
  for (int i = threadIdx.x; i < 9 * HD; i += 256) wl[i] = w2[i];
  __syncthreads();

  const int p = blockIdx.x * 256 + threadIdx.x;   // < NPIX
  const int b = p >> 14, y = (p >> 7) & (HH - 1), x = p & (WW - 1);
  float acc = b2[0];
#pragma unroll
  for (int tap = 0; tap < 9; ++tap) {
    const int dy = tap / 3 - 1, dx = tap % 3 - 1;
    const int yy = y + dy, xx = x + dx;
    if ((unsigned)yy < (unsigned)HH && (unsigned)xx < (unsigned)WW) {
      const unsigned short* a = act + ((b * HH + yy) * WW + xx) * HD;
      const float* wt = &wl[tap * HD];
#pragma unroll 8
      for (int c = 0; c < HD; c += 2)
        acc += bf2f(a[c]) * wt[c] + bf2f(a[c + 1]) * wt[c + 1];
    }
  }
  sal[p] = 1.f / (1.f + __expf(-acc));
}

// ---------------- NMS (window max) + threshold ----------------
__global__ __launch_bounds__(256)
void nms_thresh_kernel(const float* __restrict__ sal, const int* __restrict__ nr,
                       float* __restrict__ cand) {
  const int p = blockIdx.x * 256 + threadIdx.x;
  const int b = p >> 14, y = (p >> 7) & (HH - 1), x = p & (WW - 1);
  const int r = nr[0];
  const float v = sal[p];
  float m = -3.4e38f;
  for (int dy = -r; dy <= r; ++dy) {
    const int yy = y + dy;
    if ((unsigned)yy >= (unsigned)HH) continue;
    for (int dx = -r; dx <= r; ++dx) {
      const int xx = x + dx;
      if ((unsigned)xx >= (unsigned)WW) continue;
      const float s = sal[(b << 14) + (yy << 7) + xx];
      m = s > m ? s : m;
    }
  }
  cand[p] = (v == m && v > 0.1f) ? v : 0.f;
}

// ---------------- per-image iterative top-k (ties -> lower index) ----------------
__global__ __launch_bounds__(1024)
void topk_kernel(float* __restrict__ cand, float* __restrict__ kp,
                 float* __restrict__ sc, int N) {
  __shared__ float rv[1024];
  __shared__ int   ri[1024];
  const int b = blockIdx.x;
  float* c = cand + (b << 14);
  for (int i = 0; i < N; ++i) {
    float best = -2.f; int bidx = 1 << 30;
    for (int j = threadIdx.x; j < HH * WW; j += 1024) {
      const float v = c[j];
      if (v > best) { best = v; bidx = j; }
    }
    rv[threadIdx.x] = best; ri[threadIdx.x] = bidx;
    __syncthreads();
    for (int s = 512; s > 0; s >>= 1) {
      if ((int)threadIdx.x < s) {
        const float v2 = rv[threadIdx.x + s]; const int i2 = ri[threadIdx.x + s];
        if (v2 > rv[threadIdx.x] || (v2 == rv[threadIdx.x] && i2 < ri[threadIdx.x])) {
          rv[threadIdx.x] = v2; ri[threadIdx.x] = i2;
        }
      }
      __syncthreads();
    }
    if (threadIdx.x == 0) {
      const int idx = ri[0];
      kp[(b * N + i) * 2 + 0] = (float)(idx & (WW - 1));   // x
      kp[(b * N + i) * 2 + 1] = (float)(idx >> 7);         // y
      sc[b * N + i] = rv[0];
      c[idx] = -1.f;                                       // exclude
    }
    __syncthreads();
  }
}

// ---------------------------------------------------------------------------

extern "C" void kernel_launch(void* const* d_in, const int* in_sizes, int n_in,
                              void* d_out, int out_size, void* d_ws, size_t ws_size,
                              hipStream_t stream) {
  const float* dino  = (const float*)d_in[0];
  const float* w0    = (const float*)d_in[1];
  const float* b0    = (const float*)d_in[2];
  const float* g0    = (const float*)d_in[3];
  const float* beta0 = (const float*)d_in[4];
  const float* m0    = (const float*)d_in[5];
  const float* v0    = (const float*)d_in[6];
  const float* w1    = (const float*)d_in[7];
  const float* b1    = (const float*)d_in[8];
  const float* g1    = (const float*)d_in[9];
  const float* beta1 = (const float*)d_in[10];
  const float* m1    = (const float*)d_in[11];
  const float* v1    = (const float*)d_in[12];
  const float* w2    = (const float*)d_in[13];
  const float* b2    = (const float*)d_in[14];
  const int*   nr    = (const int*)d_in[16];

  float* out_sal = (float*)d_out;                       // (8,128,128,1)
  int N = (out_size - NPIX) / (BB * 3);                 // 500 for this problem
  if (N < 1) N = 1;
  float* out_kp = out_sal + NPIX;                       // (8,N,2)
  float* out_sc = out_kp + BB * N * 2;                  // (8,N)

  // workspace layout (256 B aligned)
  char* ws = (char*)d_ws;
  size_t off = 0;
  auto alloc = [&](size_t bytes) { char* p = ws + off; off = (off + bytes + 255) & ~size_t(255); return p; };
  unsigned short* x_bf  = (unsigned short*)alloc((size_t)NPIX * CIN0 * 2);
  unsigned short* act1  = (unsigned short*)alloc((size_t)NPIX * HD * 2);
  unsigned short* act2  = (unsigned short*)alloc((size_t)NPIX * HD * 2);
  unsigned short* bt0   = (unsigned short*)alloc((size_t)9 * CIN0 * HD * 2);
  unsigned short* bt1   = (unsigned short*)alloc((size_t)9 * HD * HD * 2);
  float* cand           = (float*)alloc((size_t)NPIX * 4);
  float* a0 = (float*)alloc(HD * 4);
  float* c0 = (float*)alloc(HD * 4);
  float* a1 = (float*)alloc(HD * 4);
  float* c1 = (float*)alloc(HD * 4);
  (void)ws_size; (void)n_in; (void)in_sizes;

  const int K0 = 9 * CIN0;   // 3456
  const int K1 = 9 * HD;     // 2304

  // 1) input -> bf16
  {
    int n = NPIX * CIN0;
    cvt_bf16_kernel<<<(n + 255) / 256, 256, 0, stream>>>(dino, x_bf, n);
  }
  // 2) weights -> bf16, N-major
  transpose_w_kernel<<<(K0 * HD + 255) / 256, 256, 0, stream>>>(w0, bt0, K0);
  transpose_w_kernel<<<(K1 * HD + 255) / 256, 256, 0, stream>>>(w1, bt1, K1);
  // 3) fold bias + BN
  bn_fold_kernel<<<1, HD, 0, stream>>>(b0, g0, beta0, m0, v0, a0, c0);
  bn_fold_kernel<<<1, HD, 0, stream>>>(b1, g1, beta1, m1, v1, a1, c1);
  // 4) conv0 + BN + ReLU  (WMMA implicit GEMM, K=3456)
  conv_gemm_wmma<<<NPIX / 32, 256, 0, stream>>>(x_bf, bt0, a0, c0, act1, CIN0);
  // 5) conv1 + BN + ReLU  (WMMA implicit GEMM, K=2304)
  conv_gemm_wmma<<<NPIX / 32, 256, 0, stream>>>(act1, bt1, a1, c1, act2, HD);
  // 6) conv2 + sigmoid -> saliency (f32)
  conv2_sigmoid_kernel<<<NPIX / 256, 256, 0, stream>>>(act2, w2, b2, out_sal);
  // 7) NMS + threshold
  nms_thresh_kernel<<<NPIX / 256, 256, 0, stream>>>(out_sal, nr, cand);
  // 8) per-image top-k
  topk_kernel<<<BB, 1024, 0, stream>>>(cand, out_kp, out_sc, N);
}
